// TemporalGAT_7885559955674
// MI455X (gfx1250) — matrix operand order repaired and checked by reading, not compiled, not run on hardware
//
#include <hip/hip_runtime.h>

typedef __attribute__((ext_vector_type(16))) _Float16 v16h;
typedef __attribute__((ext_vector_type(8)))  _Float16 v8h;
typedef __attribute__((ext_vector_type(8)))  float    v8f;
typedef __attribute__((ext_vector_type(4)))  unsigned int u32x4;
typedef __attribute__((ext_vector_type(8)))  int      i32x8;
typedef __attribute__((ext_vector_type(4)))  int      i32x4;

#define Nn     20000
#define Tt     64
#define Ee     640000
#define Hh     128
#define OUTC   64
#define HEADS  3
#define GATD   192      // HEADS*OUTC
#define G4     512      // 4*H
#define NEG_SLOPE 0.2f

#if defined(__has_builtin)
#if __has_builtin(__builtin_amdgcn_tensor_load_to_lds) && __has_builtin(__builtin_amdgcn_s_wait_tensorcnt)
#define HAVE_TDM 1
#endif
#if __has_builtin(__builtin_amdgcn_tanhf)
#define TANHF(x) __builtin_amdgcn_tanhf(x)        // gfx1250 v_tanh_f32
#endif
#endif
#ifndef TANHF
__device__ __forceinline__ float tanh_fast_(float x){
  float e = __expf(2.f*x);
  return (e - 1.f) * __builtin_amdgcn_rcpf(e + 1.f);
}
#define TANHF(x) tanh_fast_(x)
#endif

__device__ __forceinline__ float sigf(float x)  { return __builtin_amdgcn_rcpf(1.0f + __expf(-x)); }
__device__ __forceinline__ float lrelu(float x) { return x > 0.f ? x : NEG_SLOPE*x; }
// order-preserving float<->uint for atomic max over signed floats
__device__ __forceinline__ unsigned f2ord(float f){ unsigned u=__float_as_uint(f); return (u&0x80000000u)? ~u : (u|0x80000000u); }
__device__ __forceinline__ float ord2f(unsigned u){ return (u&0x80000000u)? __uint_as_float(u&0x7fffffffu) : __uint_as_float(~u); }

// Build 16-bit A-matrix fragment (16x32 MxK) from a 128-half LDS row.
// Contiguous K layout: lanes 0-15 hold K {0..15}, lanes 16-31 hold K {16..31} (per kc chunk).
__device__ __forceinline__ v16h load_afrag(const _Float16* rp, int kc, int khalf){
  const int kb = kc*32 + khalf*16;
  v8h lo = *(const v8h*)(rp + kb);
  v8h hi = *(const v8h*)(rp + kb + 8);
  return __builtin_shufflevector(lo, hi, 0,1,2,3,4,5,6,7,8,9,10,11,12,13,14,15);
}

// Whh [512][128] f32 -> f16, gate-interleaved rows: dst row (j*4+g) = src row (g*128+j).
// Keeps all 4 gate fragments for one hidden column within 1KB -> DS immediate offsets.
__global__ void cvt_whh_kernel(const float* __restrict__ src, _Float16* __restrict__ dst){
  int i = blockIdx.x*blockDim.x + threadIdx.x;
  if (i >= G4*Hh) return;
  int n = i >> 7, k = i & 127;          // src row n = g*128 + j
  int g = n >> 7, j = n & 127;
  dst[(unsigned)((j << 2) | g)*Hh + k] = (_Float16)src[i];
}

// ---------------------------------------------------------------------------
// LSTM: each wave owns a 16-node tile; Whh (f16, gate-interleaved) staged into
// LDS by the TDM; c-state in registers in WMMA C/D layout; h accumulated in
// packed-f16 registers per step, flushed to LDS once per t (same-wave in-order
// DS makes the next step's A-fragment loads see them without a barrier).
// ---------------------------------------------------------------------------
__global__ void __launch_bounds__(128, 1)
lstm_kernel(const float* __restrict__ xfeat,      // [N][T]
            const _Float16* __restrict__ whh16,   // [128*4][128] f16 gate-interleaved
            const float* __restrict__ Wih,        // [512][1]
            const float* __restrict__ bih,
            const float* __restrict__ bhh,
            float* __restrict__ xh)               // [N][128] relu(h_T)
{
  extern __shared__ __attribute__((aligned(32))) char smem[];
  _Float16* Wg   = (_Float16*)smem;                 // 512*128 f16 (128KB), gate-interleaved
  float*    WihS = (float*)(smem + G4*Hh*2);        // 512
  float*    bS   = WihS + G4;                       // 512
  _Float16* hS   = (_Float16*)(bS + G4);            // 4 waves * 16*128 f16
  _Float16* xS   = hS + 4*16*Hh;                    // 4 waves * 16*64 f16
  // total: 128K + 4K + 16K + 8K = 156KB  -> 2 workgroups per WGP

  const int tid = threadIdx.x, wv = tid >> 5, lane = tid & 31;

#ifdef HAVE_TDM
  if (wv == 0){
    // 2D D#: load full 512x128 f16 tile (contiguous, row stride 128) into LDS at Wg.
    unsigned long long ga = (unsigned long long)(uintptr_t)whh16;
    unsigned lds_off = (unsigned)(uintptr_t)(void*)Wg;   // LDS aperture: low 32 bits = LDS byte addr
    u32x4 g0;
    g0.x = 1u;                                           // count=1, user mode
    g0.y = lds_off;                                      // lds_addr
    g0.z = (unsigned)(ga & 0xffffffffull);               // global_addr[31:0]
    g0.w = (unsigned)((ga >> 32) & 0x1ffffffull) | (2u << 30);  // global_addr[56:32] | type=2
    i32x8 g1;
    g1[0] = (1 << 16);            // data_size=1 (2 bytes/elem)
    g1[1] = (Hh << 16);           // tensor_dim0[15:0]=128
    g1[2] = (G4 << 16);           // tensor_dim0[31:16]=0 | tensor_dim1[15:0]=512
    g1[3] = (Hh << 16);           // tensor_dim1[31:16]=0 | tile_dim0=128
    g1[4] = G4;                   // tile_dim1=512, tile_dim2=0
    g1[5] = Hh;                   // tensor_dim0_stride[31:0]=128
    g1[6] = 0;
    g1[7] = 0;
    i32x4 zg = {0,0,0,0};
#if __clang_major__ >= 23
    i32x8 zg8 = {0,0,0,0,0,0,0,0};
    __builtin_amdgcn_tensor_load_to_lds(g0, g1, zg, zg, zg8, 0);
#else
    __builtin_amdgcn_tensor_load_to_lds(g0, g1, zg, zg, 0);
#endif
    __builtin_amdgcn_s_wait_tensorcnt(0);
  }
#else
  for (int i = tid; i < G4*Hh; i += 128) Wg[i] = whh16[i];
#endif
  for (int i = tid; i < G4; i += 128){ WihS[i] = Wih[i]; bS[i] = bih[i] + bhh[i]; }

  const int node0 = blockIdx.x*64 + wv*16;
  _Float16* hw = hS + wv*(16*Hh);
  _Float16* xw = xS + wv*(16*Tt);
  for (int i = lane; i < 16*Hh; i += 32) hw[i] = (_Float16)0.f;
  for (int i = lane; i < 16*Tt; i += 32){
    int r = i >> 6, t = i & 63; int nd = node0 + r;
    xw[i] = (nd < Nn) ? (_Float16)xfeat[nd*Tt + t] : (_Float16)0.f;
  }
  __syncthreads();   // publish Wg (TDM) + tables to all waves

  const int rowA = lane & 15, khalf = lane >> 4, ncol = lane & 15;
  const v8f z = {};
  v8f cs[8];
  #pragma unroll
  for (int ch = 0; ch < 8; ++ch) cs[ch] = z;

  for (int t = 0; t < Tt; ++t){
    v16h af[4];
    #pragma unroll
    for (int kc = 0; kc < 4; ++kc) af[kc] = load_afrag(hw + rowA*Hh, kc, khalf);
    float xreg[8];                               // x[row][t], row = r + 8*khalf
    #pragma unroll
    for (int r = 0; r < 8; ++r) xreg[r] = (float)xw[(r + 8*khalf)*Tt + t];

    v8h hreg16[8];                               // packed h (f16), flushed at end of t

    #pragma unroll
    for (int ch = 0; ch < 8; ++ch){
      const int nI = ch*16 + ncol;
      // all 16 B fragments for this chunk live within 1KB of bbase -> imm offsets
      const _Float16* bbase = Wg + (unsigned)nI*(4*Hh);
      v8f acc[4];
      #pragma unroll
      for (int g = 0; g < 4; ++g) acc[g] = z;
      // flattened (kc,g) stream, B fragment software-pipelined one stage:
      // next fragment's ds_loads issue before the current WMMA -> partial dscnt wait
      v16h bcur = *(const v16h*)(bbase + khalf*16);       // kc=0, g=0
      #pragma unroll
      for (int i = 0; i < 16; ++i){
        const int kc = i >> 2, g = i & 3;
        v16h bnext = bcur;
        if (i < 15){
          const int kn = (i+1) >> 2, gn = (i+1) & 3;
          bnext = *(const v16h*)(bbase + gn*Hh + kn*32 + khalf*16);
        }
        acc[g] = __builtin_amdgcn_wmma_f32_16x16x32_f16(false, af[kc], false, bcur,
                                                        (short)0, acc[g], false, false);
        bcur = bnext;
      }
      // hoisted per-chunk gate constants (r-invariant)
      const float wI = WihS[       nI], bI = bS[       nI];
      const float wF = WihS[  Hh + nI], bF = bS[  Hh + nI];
      const float wG = WihS[2*Hh + nI], bG = bS[2*Hh + nI];
      const float wO = WihS[3*Hh + nI], bO = bS[3*Hh + nI];
      #pragma unroll
      for (int r = 0; r < 8; ++r){
        const float xi = xreg[r];
        float gi = acc[0][r] + wI*xi + bI;
        float gf = acc[1][r] + wF*xi + bF;
        float gg = acc[2][r] + wG*xi + bG;
        float go = acc[3][r] + wO*xi + bO;
        float cn = sigf(gf)*cs[ch][r] + sigf(gi)*TANHF(gg);
        float hn = sigf(go)*TANHF(cn);
        cs[ch][r] = cn;
        hreg16[ch][r] = (_Float16)hn;
      }
    }
    // flush h for this step (keeps the chunk loop store-free)
    #pragma unroll
    for (int ch = 0; ch < 8; ++ch){
      #pragma unroll
      for (int r = 0; r < 8; ++r)
        hw[(r + 8*khalf)*Hh + ch*16 + ncol] = hreg16[ch][r];
    }
  }
  // write relu(h_T) coalesced from LDS
  for (int i = lane; i < 16*Hh; i += 32){
    int r = i >> 7, c = i & 127; int nd = node0 + r;
    if (nd < Nn) xh[nd*Hh + c] = fmaxf((float)hw[i], 0.f);
  }
}

// ---------------------------------------------------------------------------
// Generic Y[N][Kout] = X[N][128] @ W[Kout][128]^T via WMMA f16
// ---------------------------------------------------------------------------
__global__ void gemm_nt_kernel(const float* __restrict__ X, const float* __restrict__ Wm,
                               float* __restrict__ Y, int nRows, int Kout)
{
  extern __shared__ __attribute__((aligned(32))) char smem[];
  _Float16* Ws = (_Float16*)smem;                  // Kout*128
  _Float16* Xs = Ws + Kout*Hh;                     // 4 waves * 16*128
  const int tid = threadIdx.x, wv = tid >> 5, lane = tid & 31;
  for (int i = tid; i < Kout*Hh; i += 128) Ws[i] = (_Float16)Wm[i];
  const int row0 = blockIdx.x*64 + wv*16;
  _Float16* Xw = Xs + wv*(16*Hh);
  for (int i = lane; i < 16*Hh; i += 32){
    int r = i >> 7, c = i & 127; int n = row0 + r;
    Xw[i] = (n < nRows) ? (_Float16)X[n*Hh + c] : (_Float16)0.f;
  }
  __syncthreads();

  const int rowA = lane & 15, khalf = lane >> 4, ncol = lane & 15;
  v16h af[4];
  #pragma unroll
  for (int kc = 0; kc < 4; ++kc) af[kc] = load_afrag(Xw + rowA*Hh, kc, khalf);

  for (int nt = 0; nt*16 < Kout; ++nt){
    v8f acc = {};
    #pragma unroll
    for (int kc = 0; kc < 4; ++kc){
      v16h bf = *(const v16h*)(Ws + (nt*16 + ncol)*Hh + kc*32 + khalf*16);
      acc = __builtin_amdgcn_wmma_f32_16x16x32_f16(false, af[kc], false, bf,
                                                   (short)0, acc, false, false);
    }
    #pragma unroll
    for (int r = 0; r < 8; ++r){
      int n = row0 + r + 8*khalf;
      if (n < nRows) Y[n*Kout + nt*16 + ncol] = acc[r];
    }
  }
}

// ------------------------- GCN edge kernels --------------------------------
__global__ void deg_init_kernel(float* deg){
  int i = blockIdx.x*blockDim.x + threadIdx.x; if (i < Nn) deg[i] = 1.0f;  // self-loop weight
}
__global__ void deg_edge_kernel(const int* __restrict__ ei, const float* __restrict__ ew, float* deg){
  int e = blockIdx.x*blockDim.x + threadIdx.x;
  if (e < Ee) atomicAdd(&deg[ei[Ee + e]], ew[e]);
}
__global__ void dinv_kernel(const float* deg, float* dinv){
  int i = blockIdx.x*blockDim.x + threadIdx.x;
  if (i < Nn){ float d = deg[i]; dinv[i] = d > 0.f ? rsqrtf(d) : 0.f; }
}
__global__ void zero_kernel(float* p, int n){
  int i = blockIdx.x*blockDim.x + threadIdx.x; if (i < n) p[i] = 0.f;
}
__global__ void gcn_scatter_kernel(const int* __restrict__ ei, const float* __restrict__ ew,
                                   const float* __restrict__ dinv, const float* __restrict__ xw,
                                   float* agg){
  long long gid = (long long)blockIdx.x*blockDim.x + threadIdx.x;
  if (gid >= (long long)Ee*32) return;
  int e = (int)(gid >> 5), c = ((int)gid & 31) * 4;     // 32 threads/edge * 4 ch
  int r = ei[e], cn = ei[Ee + e];
  __builtin_prefetch(xw + r*Hh + c, 0, 0);
  float nrm = dinv[r]*ew[e]*dinv[cn];
  float4 v = *(const float4*)(xw + r*Hh + c);
  float* dst = agg + cn*Hh + c;
  atomicAdd(dst+0, v.x*nrm); atomicAdd(dst+1, v.y*nrm);
  atomicAdd(dst+2, v.z*nrm); atomicAdd(dst+3, v.w*nrm);
}
__global__ void gcn_final_kernel(const float* agg, const float* xw, const float* dinv,
                                 const float* b, float* out){
  int gid = blockIdx.x*blockDim.x + threadIdx.x;
  if (gid >= Nn*Hh) return;
  int n = gid >> 7, c = gid & 127;
  float di = dinv[n];
  out[gid] = fmaxf(agg[gid] + xw[gid]*di*di + b[c], 0.f);  // self-loop + bias + relu
}

// ------------------------- GAT kernels -------------------------------------
__global__ void gat_att_kernel(const float* __restrict__ xg, const float* __restrict__ aS,
                               const float* __restrict__ aD, float* a_src, float* a_dst){
  int gid = blockIdx.x*blockDim.x + threadIdx.x;
  if (gid >= Nn*HEADS) return;
  int n = gid / HEADS, hh = gid % HEADS;
  const float* xr = xg + n*GATD + hh*OUTC;
  float s = 0.f, d = 0.f;
  for (int c = 0; c < OUTC; ++c){ float v = xr[c]; s += v*aS[hh*OUTC+c]; d += v*aD[hh*OUTC+c]; }
  a_src[gid] = s; a_dst[gid] = d;
}
__global__ void amax_init_kernel(const float* a_src, const float* a_dst, unsigned* amax){
  int gid = blockIdx.x*blockDim.x + threadIdx.x;
  if (gid < Nn*HEADS) amax[gid] = f2ord(lrelu(a_src[gid] + a_dst[gid]));  // self-loop alpha
}
__global__ void amax_edge_kernel(const int* __restrict__ ei, const float* a_src,
                                 const float* a_dst, unsigned* amax){
  long long gid = (long long)blockIdx.x*blockDim.x + threadIdx.x;
  if (gid >= (long long)Ee*HEADS) return;
  int e = (int)(gid / HEADS), hh = (int)(gid % HEADS);
  int r = ei[e], cn = ei[Ee + e];
  atomicMax(&amax[cn*HEADS + hh], f2ord(lrelu(a_src[r*HEADS+hh] + a_dst[cn*HEADS+hh])));
}
__global__ void denom_init_kernel(const float* a_src, const float* a_dst,
                                  const unsigned* amax, float* denom){
  int gid = blockIdx.x*blockDim.x + threadIdx.x;
  if (gid >= Nn*HEADS) return;
  denom[gid] = __expf(lrelu(a_src[gid] + a_dst[gid]) - ord2f(amax[gid]));
}
__global__ void denom_edge_kernel(const int* __restrict__ ei, const float* a_src,
                                  const float* a_dst, const unsigned* amax, float* denom){
  long long gid = (long long)blockIdx.x*blockDim.x + threadIdx.x;
  if (gid >= (long long)Ee*HEADS) return;
  int e = (int)(gid / HEADS), hh = (int)(gid % HEADS);
  int r = ei[e], cn = ei[Ee + e]; int idx = cn*HEADS + hh;
  atomicAdd(&denom[idx], __expf(lrelu(a_src[r*HEADS+hh] + a_dst[idx]) - ord2f(amax[idx])));
}
__global__ void gat_init_kernel(const float* xg, const float* a_src, const float* a_dst,
                                const unsigned* amax, const float* denom, float* gagg){
  int gid = blockIdx.x*blockDim.x + threadIdx.x;
  if (gid >= Nn*GATD) return;
  int n = gid / GATD, c = gid % GATD, hh = c / OUTC, idx = n*HEADS + hh;
  float w = __expf(lrelu(a_src[idx] + a_dst[idx]) - ord2f(amax[idx])) / denom[idx];
  gagg[gid] = w * xg[gid];                                    // self-loop term (init, not +=)
}
__global__ void gat_edge_kernel(const int* __restrict__ ei, const float* __restrict__ xg,
                                const float* a_src, const float* a_dst,
                                const unsigned* amax, const float* denom, float* gagg){
  long long gid = (long long)blockIdx.x*blockDim.x + threadIdx.x;
  if (gid >= (long long)Ee*48) return;
  int e = (int)(gid / 48), rr = (int)(gid % 48);
  int hh = rr >> 4, c = (rr & 15) * 4;                        // 48 threads/edge * 4 ch
  int r = ei[e], cn = ei[Ee + e]; int idx = cn*HEADS + hh;
  __builtin_prefetch(xg + r*GATD + hh*OUTC + c, 0, 0);
  float w = __expf(lrelu(a_src[r*HEADS+hh] + a_dst[idx]) - ord2f(amax[idx])) / denom[idx];
  float4 v = *(const float4*)(xg + r*GATD + hh*OUTC + c);
  float* dst = gagg + cn*GATD + hh*OUTC + c;
  atomicAdd(dst+0, w*v.x); atomicAdd(dst+1, w*v.y);
  atomicAdd(dst+2, w*v.z); atomicAdd(dst+3, w*v.w);
}
// 64 partial groups per column -> 12K strided partial sums, 1 atomic each
__global__ void mean_accum_kernel(const float* __restrict__ gagg, float* res){
  int gid = blockIdx.x*blockDim.x + threadIdx.x;
  if (gid >= GATD*64) return;
  int c = gid % GATD, grp = gid / GATD;
  float s = 0.f;
  for (int n = grp; n < Nn; n += 64) s += gagg[n*GATD + c];
  atomicAdd(&res[c], s);
}
__global__ void final_kernel(const float* res, const float* b, float* out){
  int c = threadIdx.x;
  if (c < GATD) out[c] = res[c] * (1.0f/(float)Nn) + b[c];
}

// ---------------------------------------------------------------------------
extern "C" void kernel_launch(void* const* d_in, const int* in_sizes, int n_in,
                              void* d_out, int out_size, void* d_ws, size_t ws_size,
                              hipStream_t stream)
{
  (void)in_sizes; (void)n_in; (void)out_size; (void)ws_size;
  const float* xfeat = (const float*)d_in[0];
  const int*   ei    = (const int*)  d_in[1];
  const float* ew    = (const float*)d_in[2];
  const float* Wih   = (const float*)d_in[3];
  const float* Whh   = (const float*)d_in[4];
  const float* bih   = (const float*)d_in[5];
  const float* bhh   = (const float*)d_in[6];
  const float* gcnW  = (const float*)d_in[7];
  const float* gcnB  = (const float*)d_in[8];
  const float* gatW  = (const float*)d_in[9];
  const float* attS  = (const float*)d_in[10];
  const float* attD  = (const float*)d_in[11];
  const float* gatB  = (const float*)d_in[12];
  float* out = (float*)d_out;

  float* ws    = (float*)d_ws;
  float* xh    = ws;  ws += Nn*Hh;       // relu(lstm h)
  float* xw    = ws;  ws += Nn*Hh;       // xh @ gcnW^T
  float* agg   = ws;  ws += Nn*Hh;       // gcn edge accumulation
  float* gcno  = ws;  ws += Nn*Hh;       // relu(gcn out)
  float* xg    = ws;  ws += Nn*GATD;     // gcno @ gatW^T
  float* gagg  = ws;  ws += Nn*GATD;     // gat aggregation
  float* deg   = ws;  ws += Nn;
  float* dinv  = ws;  ws += Nn;
  float* a_src = ws;  ws += Nn*HEADS;
  float* a_dst = ws;  ws += Nn*HEADS;
  float* denom = ws;  ws += Nn*HEADS;
  float* res   = ws;  ws += GATD;
  unsigned* amax = (unsigned*)ws; ws += Nn*HEADS;
  _Float16* whh16 = (_Float16*)ws;       // 512*128 halves (gate-interleaved)

  (void)hipFuncSetAttribute((const void*)lstm_kernel,
        hipFuncAttributeMaxDynamicSharedMemorySize, 160*1024);
  (void)hipFuncSetAttribute((const void*)gemm_nt_kernel,
        hipFuncAttributeMaxDynamicSharedMemorySize, 96*1024);

  const dim3 b256(256);
  const int NB = (Nn + 63) / 64;

  cvt_whh_kernel<<<(G4*Hh+255)/256, b256, 0, stream>>>(Whh, whh16);

  size_t lstm_smem = (size_t)G4*Hh*2 + G4*8 + (size_t)4*16*Hh*2 + (size_t)4*16*Tt*2;
  lstm_kernel<<<NB, 128, lstm_smem, stream>>>(xfeat, whh16, Wih, bih, bhh, xh);

  size_t g128_smem = (size_t)Hh*Hh*2 + (size_t)4*16*Hh*2;
  gemm_nt_kernel<<<NB, 128, g128_smem, stream>>>(xh, gcnW, xw, Nn, Hh);

  deg_init_kernel<<<(Nn+255)/256, b256, 0, stream>>>(deg);
  deg_edge_kernel<<<(Ee+255)/256, b256, 0, stream>>>(ei, ew, deg);
  dinv_kernel<<<(Nn+255)/256, b256, 0, stream>>>(deg, dinv);
  zero_kernel<<<(Nn*Hh+255)/256, b256, 0, stream>>>(agg, Nn*Hh);
  gcn_scatter_kernel<<<(unsigned)(((long long)Ee*32 + 255)/256), b256, 0, stream>>>(ei, ew, dinv, xw, agg);
  gcn_final_kernel<<<(Nn*Hh+255)/256, b256, 0, stream>>>(agg, xw, dinv, gcnB, gcno);

  size_t g192_smem = (size_t)GATD*Hh*2 + (size_t)4*16*Hh*2;
  gemm_nt_kernel<<<NB, 128, g192_smem, stream>>>(gcno, gatW, xg, Nn, GATD);

  gat_att_kernel<<<(Nn*HEADS+255)/256, b256, 0, stream>>>(xg, attS, attD, a_src, a_dst);
  amax_init_kernel<<<(Nn*HEADS+255)/256, b256, 0, stream>>>(a_src, a_dst, amax);
  amax_edge_kernel<<<(unsigned)(((long long)Ee*HEADS + 255)/256), b256, 0, stream>>>(ei, a_src, a_dst, amax);
  denom_init_kernel<<<(Nn*HEADS+255)/256, b256, 0, stream>>>(a_src, a_dst, amax, denom);
  denom_edge_kernel<<<(unsigned)(((long long)Ee*HEADS + 255)/256), b256, 0, stream>>>(ei, a_src, a_dst, amax, denom);
  gat_init_kernel<<<(Nn*GATD+255)/256, b256, 0, stream>>>(xg, a_src, a_dst, amax, denom, gagg);
  gat_edge_kernel<<<(unsigned)(((long long)Ee*48 + 255)/256), b256, 0, stream>>>(ei, xg, a_src, a_dst, amax, denom, gagg);
  zero_kernel<<<1, b256, 0, stream>>>(res, GATD);
  mean_accum_kernel<<<(GATD*64+255)/256, b256, 0, stream>>>(gagg, res);
  final_kernel<<<1, b256, 0, stream>>>(res, gatB, out);
}
